// BondConvLayer_19078244729259
// MI455X (gfx1250) — matrix operand order repaired and checked by reading, not compiled, not run on hardware
//
#include <hip/hip_runtime.h>
#include <hip/hip_bf16.h>

// ---------------- problem constants ----------------
#define ATOM_F   64
#define NBR_F    64
#define ANG_TOT  1056          // 66 * 16
#define ANG_U    (ANG_TOT/2)   // 528 u32 per packed angle row
#define TOTAL_F  1312          // 64 + 64 + 64 + 64 + 1056
#define HDIM     128
#define OUT_F    64
#define MNBR     12
#define BN_EPS   1e-5f

// W1 column blocks: [0,64)=center  [64,128)=nbr_atom  [128,192)=nbr  [192,256)=rolled  [256,1312)=angle
#define KW1_U    (TOTAL_F/2)   // 656 u32 per W1 row (bf16 pairs)
#define KW2_U    (HDIM/2)      // 64  u32 per W2 row

// Angle K streamed in chunks of 96 (3 WMMA k-steps per chunk), 11 chunks.
#define ANG_CHUNK    96
#define ANG_CHUNK_U  48        // u32 per row per chunk
#define N_CHUNKS     11

// CDNA5 async global->LDS path (ASYNCcnt), with synchronous fallback.
#if defined(__AMDGCN__) && __has_builtin(__builtin_amdgcn_global_load_async_to_lds_b128) && __has_builtin(__builtin_amdgcn_s_wait_asynccnt)
#define HAVE_ASYNC 1
#else
#define HAVE_ASYNC 0
#endif

// ---------------- bf16 helpers (bit-exact RNE) ----------------
__device__ __forceinline__ unsigned short f2bf(float f) {
  unsigned u = __float_as_uint(f);
  unsigned r = u + 0x7FFFu + ((u >> 16) & 1u);
  return (unsigned short)(r >> 16);
}
__device__ __forceinline__ unsigned pack2(float f0, float f1) {
  return (unsigned)f2bf(f0) | ((unsigned)f2bf(f1) << 16);
}
__device__ __forceinline__ float bf2f(unsigned short s) {
  return __uint_as_float(((unsigned)s) << 16);
}
__device__ __forceinline__ float softplus_f(float x) {
  return fmaxf(x, 0.f) + log1pf(expf(-fabsf(x)));
}

// ---------------- WMMA fragment plumbing (CDNA5 16x16x32 bf16, wave32) ------
typedef __attribute__((ext_vector_type(16))) __bf16 v16bf;
typedef __attribute__((ext_vector_type(8)))  float  v8f;
union ABfrag { v16bf v; unsigned u[8]; };

// A (16x32, MxK) contiguous k order (k = 16*hi + s); sm = u32 array of bf16 k-pairs.
__device__ __forceinline__ v16bf load_a(const unsigned* sm, int strideU,
                                        int row, int kbU, int hi) {
  ABfrag a;
  int idx = row * strideU + kbU + hi * 8;
#pragma unroll
  for (int j = 0; j < 8; ++j) a.u[j] = sm[idx + j];
  return a.v;
}
// B (32x16, KxN): column N = row `col` of the row-major weight matrix.
__device__ __forceinline__ v16bf load_b(const unsigned* W, int KwU,
                                        int col, int kbU, int hi) {
  ABfrag b;
  int idx = col * KwU + kbU + hi * 8;
#pragma unroll
  for (int j = 0; j < 8; ++j) b.u[j] = W[idx + j];
  return b.v;
}
__device__ __forceinline__ v8f wmma_bf16(v16bf a, v16bf b, v8f c) {
  return __builtin_amdgcn_wmma_f32_16x16x32_bf16(false, a, false, b,
                                                 (short)0, c, false, false);
}

#if HAVE_ASYNC
// Builtin prototype (per hipcc diagnostic): (int4*, int4*, imm int, imm int)
typedef __attribute__((__vector_size__(4 * sizeof(int)))) int i32x4;
// One wave moves a 16-row x 96-k bf16 angle chunk with 6 b128 async copies;
// each active lane transfers 16 bytes (row = lane&15, half = lane>>4).
__device__ __forceinline__ void issue_angle_chunk(const unsigned* angU,
                                                  unsigned* dst,
                                                  int a0, int kb, int lane) {
  int row = lane & 15, half = lane >> 4;
  const unsigned* g = angU + (size_t)(a0 + row) * ANG_U + (kb >> 1);
  unsigned* l = dst + row * ANG_CHUNK_U;
#pragma unroll
  for (int i = 0; i < 6; ++i) {
    int seg = half + 2 * i;                 // 12 x 16B segments per row
    __builtin_amdgcn_global_load_async_to_lds_b128(
        (i32x4*)(g + seg * 4), (i32x4*)(l + seg * 4), 0, 0);
  }
}
#endif

// ---------------- K0: pack weights to bf16 pairs, zero BN accumulators -------
__global__ void k0_prep(const float* __restrict__ W1, const float* __restrict__ W2,
                        unsigned* __restrict__ W1u, unsigned* __restrict__ W2u,
                        float* __restrict__ st1, float* __restrict__ st2) {
  int t = blockIdx.x * 256 + threadIdx.x;
  if (t < (HDIM * TOTAL_F) / 2) W1u[t] = pack2(W1[2 * t], W1[2 * t + 1]);
  if (t < (OUT_F * HDIM) / 2)   W2u[t] = pack2(W2[2 * t], W2[2 * t + 1]);
  if (t < 2 * HDIM)  st1[t] = 0.f;   // sum[128] + sumsq[128]
  if (t < 2 * OUT_F) st2[t] = 0.f;   // sum[64]  + sumsq[64]
}

// ---------------- K0b: pack angle features to bf16 pairs ---------------------
__global__ void k0b_angle_pack(const float* __restrict__ ang,
                               unsigned* __restrict__ angU, int totalU) {
  int t = blockIdx.x * 256 + threadIdx.x;
  if (t < totalU) angU[t] = pack2(ang[2 * t], ang[2 * t + 1]);
}

// ---------------- K1: per-atom S[n,h], T[n,h]  (factorized big GEMM) ---------
// Block: 256 threads = 8 waves; tile = 16 atoms x 128 h; one wave per 16 cols.
// Angle K (1056) streamed via double-buffered async global->LDS copies.
__global__ void k1_atom_gemm(const float* __restrict__ atom,
                             const unsigned* __restrict__ angU,
                             const float* __restrict__ b1,
                             const unsigned* __restrict__ W1u,
                             float* __restrict__ S, float* __restrict__ T) {
  __shared__ unsigned sAtomU[16 * 32];                       // 16 x 64 bf16
  __shared__ __align__(16) unsigned sAngU[2][16 * ANG_CHUNK_U];
  const int t = threadIdx.x, wv = t >> 5, lane = t & 31;
  const int colL = lane & 15, hi = lane >> 4;
  const int a0 = blockIdx.x * 16;

  for (int i = t; i < 512; i += 256) {
    int r = i >> 5, kk = i & 31;
    const float* p = atom + (a0 + r) * ATOM_F + 2 * kk;
    sAtomU[i] = pack2(p[0], p[1]);
  }
#if HAVE_ASYNC
  if (wv == 0) issue_angle_chunk(angU, &sAngU[0][0], a0, 0, lane);
#endif
  __syncthreads();

  const int col = wv * 16 + colL;
  v8f cS = {}; v8f cT = {};
#pragma unroll
  for (int kb = 0; kb < ATOM_F; kb += 32) {
    v16bf a  = load_a(sAtomU, 32, colL, kb / 2, hi);
    v16bf bS = load_b(W1u, KW1_U, col, (0 + kb) / 2, hi);     // center block
    cS = wmma_bf16(a, bS, cS);
    v16bf bT = load_b(W1u, KW1_U, col, (64 + kb) / 2, hi);    // nbr-atom block
    cT = wmma_bf16(a, bT, cT);
  }

  for (int c = 0; c < N_CHUNKS; ++c) {
    const int kb = c * ANG_CHUNK;
#if HAVE_ASYNC
    if (c < N_CHUNKS - 1) {
      if (wv == 0) issue_angle_chunk(angU, &sAngU[(c + 1) & 1][0],
                                     a0, kb + ANG_CHUNK, lane);
      __builtin_amdgcn_s_wait_asynccnt(6);   // in-order: chunk c is resident
    } else {
      __builtin_amdgcn_s_wait_asynccnt(0);
    }
    __syncthreads();
    const unsigned* buf = &sAngU[c & 1][0];
#else
    __syncthreads();
    for (int i = t; i < 16 * ANG_CHUNK_U; i += 256) {
      int r = i / ANG_CHUNK_U, kk = i % ANG_CHUNK_U;
      sAngU[0][i] = angU[(size_t)(a0 + r) * ANG_U + kb / 2 + kk];
    }
    __syncthreads();
    const unsigned* buf = &sAngU[0][0];
#endif
#pragma unroll
    for (int kk = 0; kk < ANG_CHUNK; kk += 32) {
      v16bf a = load_a(buf, ANG_CHUNK_U, colL, kk / 2, hi);
      v16bf b = load_b(W1u, KW1_U, col, (256 + kb + kk) / 2, hi);  // angle block
      cS = wmma_bf16(a, b, cS);
    }
#if HAVE_ASYNC
    __syncthreads();   // all reads done before async engine overwrites buffer
#endif
  }

  const float bias = b1[col];
  const int rowBase = a0 + hi * 8;
#pragma unroll
  for (int r = 0; r < 8; ++r) {
    S[(rowBase + r) * HDIM + col] = cS[r] + bias;
    T[(rowBase + r) * HDIM + col] = cT[r];
  }
}

// ---------------- K2: per-row x1 = S + gather(T) + nbr·Wc + roll(nbr)·Wd -----
// Block: 256 threads; tile = 16 rows x 128 h. Also accumulates BN1 stats.
__global__ void k2_pair_gemm(const float* __restrict__ nbr,
                             const int* __restrict__ idx,
                             const unsigned* __restrict__ W1u,
                             const float* __restrict__ S,
                             const float* __restrict__ T,
                             unsigned short* __restrict__ X1,
                             float* __restrict__ gsum1, float* __restrict__ gsq1) {
  __shared__ unsigned sA1U[16 * 32], sA2U[16 * 32];
  __shared__ float ssum[HDIM], ssq[HDIM];
  const int t = threadIdx.x, wv = t >> 5, lane = t & 31;
  const int colL = lane & 15, hi = lane >> 4;
  const int r0 = blockIdx.x * 16;
  if (t < HDIM) { ssum[t] = 0.f; ssq[t] = 0.f; }

  for (int i = t; i < 512; i += 256) {
    int r = i >> 5, kk = i & 31;
    int grow = r0 + r;
    const float* p1 = nbr + grow * NBR_F + 2 * kk;
    sA1U[i] = pack2(p1[0], p1[1]);
    int n = grow / MNBR, m = grow - n * MNBR;
    int m2 = m + 1; if (m2 == MNBR) m2 = 0;                   // jnp.roll(-1, axis=1)
    const float* p2 = nbr + (n * MNBR + m2) * NBR_F + 2 * kk;
    sA2U[i] = pack2(p2[0], p2[1]);
  }
  __syncthreads();

  const int col = wv * 16 + colL;
  v8f c = {};
#pragma unroll
  for (int kb = 0; kb < NBR_F; kb += 32) {
    v16bf a1 = load_a(sA1U, 32, colL, kb / 2, hi);
    v16bf bc = load_b(W1u, KW1_U, col, (128 + kb) / 2, hi);   // nbr block
    c = wmma_bf16(a1, bc, c);
    v16bf a2 = load_a(sA2U, 32, colL, kb / 2, hi);
    v16bf bd = load_b(W1u, KW1_U, col, (192 + kb) / 2, hi);   // rolled block
    c = wmma_bf16(a2, bd, c);
  }
  float psum = 0.f, psq = 0.f;
  const int rowBase = r0 + hi * 8;
#pragma unroll
  for (int r = 0; r < 8; ++r) {
    int grow = rowBase + r;
    int n = grow / MNBR;
    int j = idx[grow];
    float v = c[r] + S[n * HDIM + col] + T[j * HDIM + col];
    psum += v; psq += v * v;
    X1[grow * HDIM + col] = f2bf(v);
  }
  atomicAdd(&ssum[col], psum);
  atomicAdd(&ssq[col], psq);
  __syncthreads();
  if (t < HDIM) { atomicAdd(&gsum1[t], ssum[t]); atomicAdd(&gsq1[t], ssq[t]); }
}

// ---------------- BN finalize: bn[h]=scale, bn[h+Hd]=shift ------------------
__global__ void k_bnfinalize(const float* __restrict__ gsum, const float* __restrict__ gsq,
                             const float* __restrict__ g, const float* __restrict__ be,
                             float cntInv, int Hd, float* __restrict__ bn) {
  int t = threadIdx.x;
  if (t < Hd) {
    float mean = gsum[t] * cntInv;
    float var  = fmaxf(gsq[t] * cntInv - mean * mean, 0.f);
    float sc   = g[t] * rsqrtf(var + BN_EPS);
    bn[t] = sc;
    bn[t + Hd] = be[t] - mean * sc;
  }
}

// ---------------- K4: BN1+softplus fused -> GEMM2 (K=128) -> Z (=d_out) -----
// Block: 256 threads; tile = 32 rows x 64 cols (8 waves: 2 row-tiles x 4 col-tiles).
__global__ void k4_act_gemm2(const unsigned short* __restrict__ X1,
                             const float* __restrict__ bn1,
                             const unsigned* __restrict__ W2u,
                             const float* __restrict__ b2,
                             float* __restrict__ Z,
                             float* __restrict__ gsum2, float* __restrict__ gsq2) {
  __shared__ unsigned sYU[32 * 64];   // 32 rows x 128 bf16 (64 u32)
  __shared__ float ssum[OUT_F], ssq[OUT_F];
  const int t = threadIdx.x, wv = t >> 5, lane = t & 31;
  const int colL = lane & 15, hi = lane >> 4;
  const int r0 = blockIdx.x * 32;
  if (t < OUT_F) { ssum[t] = 0.f; ssq[t] = 0.f; }
  const float* scale1 = bn1;
  const float* shift1 = bn1 + HDIM;

  for (int i = t; i < 2048; i += 256) {
    int r = i >> 6, kk = i & 63, k0 = 2 * kk;
    int grow = r0 + r;
    float x0 = bf2f(X1[grow * HDIM + k0]);
    float x1 = bf2f(X1[grow * HDIM + k0 + 1]);
    float v0 = softplus_f(x0 * scale1[k0]     + shift1[k0]);
    float v1 = softplus_f(x1 * scale1[k0 + 1] + shift1[k0 + 1]);
    sYU[i] = pack2(v0, v1);
  }
  __syncthreads();

  const int rowT = (wv >> 2) * 16;
  const int col  = (wv & 3) * 16 + colL;
  v8f c = {};
#pragma unroll
  for (int kb = 0; kb < HDIM; kb += 32) {
    v16bf a = load_a(sYU + rowT * 64, 64, colL, kb / 2, hi);
    v16bf b = load_b(W2u, KW2_U, col, kb / 2, hi);
    c = wmma_bf16(a, b, c);
  }
  const float bias = b2[col];
  float psum = 0.f, psq = 0.f;
  const int rowBase = r0 + rowT + hi * 8;
#pragma unroll
  for (int r = 0; r < 8; ++r) {
    float v = c[r] + bias;
    Z[(rowBase + r) * OUT_F + col] = v;       // pre-BN2 activations into d_out
    psum += v; psq += v * v;
  }
  atomicAdd(&ssum[col], psum);
  atomicAdd(&ssq[col], psq);
  __syncthreads();
  if (t < OUT_F) { atomicAdd(&gsum2[t], ssum[t]); atomicAdd(&gsq2[t], ssq[t]); }
}

// ---------------- K6: BN2 + softplus + bond-weight, in place on d_out -------
__global__ void k6_final(float* __restrict__ Z, const float* __restrict__ bn2,
                         const float* __restrict__ bw, int total) {
  int e = blockIdx.x * 256 + threadIdx.x;
  if (e < total) {
    int c = e & (OUT_F - 1), row = e >> 6;
    float v = Z[e] * bn2[c] + bn2[OUT_F + c];
    Z[e] = softplus_f(v) * bw[row];
  }
}

// ---------------- launch ----------------------------------------------------
extern "C" void kernel_launch(void* const* d_in, const int* in_sizes, int n_in,
                              void* d_out, int out_size, void* d_ws, size_t ws_size,
                              hipStream_t stream) {
  const float* atom  = (const float*)d_in[0];
  const float* nbr   = (const float*)d_in[1];
  const int*   idx   = (const int*)  d_in[2];
  const float* angle = (const float*)d_in[3];
  const float* bw    = (const float*)d_in[4];
  const float* W1    = (const float*)d_in[5];
  const float* b1    = (const float*)d_in[6];
  const float* g1    = (const float*)d_in[7];
  const float* be1   = (const float*)d_in[8];
  const float* W2    = (const float*)d_in[9];
  const float* b2    = (const float*)d_in[10];
  const float* g2    = (const float*)d_in[11];
  const float* be2   = (const float*)d_in[12];
  float* out = (float*)d_out;

  const int nAtoms = in_sizes[0] / ATOM_F;         // 20000
  const int R      = in_sizes[1] / NBR_F;          // 240000 rows

  // workspace layout (bytes; every slot size is a multiple of 256)
  char* ws = (char*)d_ws;
  size_t o = 0;
  unsigned* W1u = (unsigned*)(ws + o);       o += (size_t)(HDIM * TOTAL_F / 2) * 4;   // 335,872
  unsigned* W2u = (unsigned*)(ws + o);       o += (size_t)(OUT_F * HDIM / 2) * 4;     //  16,384
  unsigned* angU = (unsigned*)(ws + o);      o += (size_t)nAtoms * ANG_U * 4;         //  42.2 MB
  float* S      = (float*)(ws + o);          o += (size_t)nAtoms * HDIM * 4;
  float* T      = (float*)(ws + o);          o += (size_t)nAtoms * HDIM * 4;
  unsigned short* X1 = (unsigned short*)(ws + o); o += (size_t)R * HDIM * 2;
  float* st1    = (float*)(ws + o);          o += 2 * HDIM * 4;                        // sum,sumsq
  float* st2    = (float*)(ws + o);          o += 2 * OUT_F * 4;
  float* bn1    = (float*)(ws + o);          o += 2 * HDIM * 4;                        // scale,shift
  float* bn2    = (float*)(ws + o);          o += 2 * OUT_F * 4;
  (void)ws_size; (void)n_in; (void)out_size;

  const int prepBlocks = (HDIM * TOTAL_F / 2 + 255) / 256;   // 328
  k0_prep<<<prepBlocks, 256, 0, stream>>>(W1, W2, W1u, W2u, st1, st2);

  const int angTotU = nAtoms * ANG_U;
  k0b_angle_pack<<<(angTotU + 255) / 256, 256, 0, stream>>>(angle, angU, angTotU);

  k1_atom_gemm<<<nAtoms / 16, 256, 0, stream>>>(atom, angU, b1, W1u, S, T);

  k2_pair_gemm<<<R / 16, 256, 0, stream>>>(nbr, idx, W1u, S, T, X1,
                                           st1, st1 + HDIM);

  k_bnfinalize<<<1, HDIM, 0, stream>>>(st1, st1 + HDIM, g1, be1,
                                       1.0f / (float)R, HDIM, bn1);

  k4_act_gemm2<<<R / 32, 256, 0, stream>>>(X1, bn1, W2u, b2, out,
                                           st2, st2 + OUT_F);

  k_bnfinalize<<<1, OUT_F, 0, stream>>>(st2, st2 + OUT_F, g2, be2,
                                        1.0f / (float)R, OUT_F, bn2);

  const int total = R * OUT_F;
  k6_final<<<(total + 255) / 256, 256, 0, stream>>>(out, bn2, bw, total);
}
